// MMDLoss_56203942035788
// MI455X (gfx1250) — matrix operand order repaired and checked by reading, not compiled, not run on hardware
//
#include <hip/hip_runtime.h>
#include <hip/hip_bf16.h>
#include <math.h>

// Problem constants (from reference): B=8, NC=64, NO=576, D=768
#define NX    512            // n = B*NC_TOK
#define NY    4608           // m = B*NO_TOK
#define NTOT  5120           // N = n + m
#define DDIM  768
#define TILES 320            // NTOT / 16
#define PTILES (TILES * (TILES / 2))   // wave computes a 16x32 pair-tile
#define WPB   8              // waves per block (256 threads, wave32)
#define NBINS 4096
#define KNUM  5

typedef __attribute__((ext_vector_type(16))) __bf16 v16bf;
typedef __attribute__((ext_vector_type(8)))  float  v8f;

union FragBF { uint4 u[2]; v16bf v; };
struct Tile2 { v8f c0, c1; };

// ---------------- fp32 <-> bf16 helpers ----------------
__device__ __forceinline__ unsigned short f2bf(float f) {
    unsigned int u = __float_as_uint(f);
    u += 0x7FFFu + ((u >> 16) & 1u);     // round-to-nearest-even
    return (unsigned short)(u >> 16);
}
__device__ __forceinline__ float bf2f(unsigned short b) {
    return __uint_as_float(((unsigned int)b) << 16);
}

// ---------------- init: zero accumulators ----------------
__global__ void init_kernel(unsigned* h1, unsigned* h2, double* sums, unsigned* scal) {
    for (int i = threadIdx.x; i < NBINS; i += blockDim.x) { h1[i] = 0u; h2[i] = 0u; }
    if (threadIdx.x < 3) sums[threadIdx.x] = 0.0;
    if (threadIdx.x < 8) scal[threadIdx.x] = 0u;
}

// ---------------- prep: bf16 hi/lo split + row sq-norms + max norm ----------------
__global__ void prep_kernel(const float* __restrict__ X, const float* __restrict__ Y,
                            unsigned short* __restrict__ hi, unsigned short* __restrict__ lo,
                            float* __restrict__ sq, unsigned* __restrict__ scal_u) {
    __shared__ float red[256];
    const int r = blockIdx.x;
    const float* src = (r < NX) ? (X + (size_t)r * DDIM) : (Y + (size_t)(r - NX) * DDIM);
    float ss = 0.f;
    for (int d = threadIdx.x; d < DDIM; d += blockDim.x) {
        float x = src[d];
        unsigned short hb = f2bf(x);
        unsigned short lb = f2bf(x - bf2f(hb));
        hi[(size_t)r * DDIM + d] = hb;
        lo[(size_t)r * DDIM + d] = lb;
        ss += x * x;
    }
    red[threadIdx.x] = ss;
    __syncthreads();
    for (int s = 128; s > 0; s >>= 1) {
        if (threadIdx.x < s) red[threadIdx.x] += red[threadIdx.x + s];
        __syncthreads();
    }
    if (threadIdx.x == 0) {
        sq[r] = red[0];
        atomicMax(&scal_u[0], __float_as_uint(red[0]));   // floats >= 0: uint order == float order
    }
}

// ---------------- 16x32 Gram pair-tile via bf16x3 WMMA ----------------
// One wave computes two adjacent 16x16 output tiles (j-tiles jbase0 and jbase0+16),
// reusing the A fragments across both, with 4 independent accumulator chains:
//   chh* : hi*hi      cmx* : hi*lo + lo*hi     (dot ~= chh + cmx, near-fp32)
// A frag (16x32 bf16): lane m = lane&15, K-half h = lane>>4 -> 16 contiguous elems.
// B frag (32x16 bf16): lane n = lane&15, K-half h -> 16 contiguous elems.
__device__ __forceinline__ Tile2 gram_tile2(const unsigned short* __restrict__ hi,
                                            const unsigned short* __restrict__ lo,
                                            int ibase, int jbase0, int lane) {
    const int mn = lane & 15;
    const int h  = lane >> 4;
    const unsigned short* aH  = hi + (size_t)(ibase + mn) * DDIM;
    const unsigned short* aL  = lo + (size_t)(ibase + mn) * DDIM;
    const unsigned short* b0H = hi + (size_t)(jbase0 + mn) * DDIM;
    const unsigned short* b0L = lo + (size_t)(jbase0 + mn) * DDIM;
    const unsigned short* b1H = b0H + (size_t)16 * DDIM;
    const unsigned short* b1L = b0L + (size_t)16 * DDIM;

    v8f chh0 = {0.f,0.f,0.f,0.f,0.f,0.f,0.f,0.f};
    v8f cmx0 = {0.f,0.f,0.f,0.f,0.f,0.f,0.f,0.f};
    v8f chh1 = {0.f,0.f,0.f,0.f,0.f,0.f,0.f,0.f};
    v8f cmx1 = {0.f,0.f,0.f,0.f,0.f,0.f,0.f,0.f};

    for (int ks = 0; ks < DDIM; ks += 32) {
        FragBF ahi, alo, b0hi, b0lo, b1hi, b1lo;
        const unsigned short* pa  = aH + ks + h * 16;
        const unsigned short* pal = aL + ks + h * 16;
        ahi.u[0] = *(const uint4*)(pa);         ahi.u[1] = *(const uint4*)(pa + 8);
        alo.u[0] = *(const uint4*)(pal);        alo.u[1] = *(const uint4*)(pal + 8);
        const int bo = ks + h * 16;
        b0hi.u[0] = *(const uint4*)(b0H + bo);  b0hi.u[1] = *(const uint4*)(b0H + bo + 8);
        b0lo.u[0] = *(const uint4*)(b0L + bo);  b0lo.u[1] = *(const uint4*)(b0L + bo + 8);
        b1hi.u[0] = *(const uint4*)(b1H + bo);  b1hi.u[1] = *(const uint4*)(b1H + bo + 8);
        b1lo.u[0] = *(const uint4*)(b1L + bo);  b1lo.u[1] = *(const uint4*)(b1L + bo + 8);

        chh0 = __builtin_amdgcn_wmma_f32_16x16x32_bf16(false, ahi.v, false, b0hi.v, (short)0, chh0, false, false);
        chh1 = __builtin_amdgcn_wmma_f32_16x16x32_bf16(false, ahi.v, false, b1hi.v, (short)0, chh1, false, false);
        cmx0 = __builtin_amdgcn_wmma_f32_16x16x32_bf16(false, ahi.v, false, b0lo.v, (short)0, cmx0, false, false);
        cmx1 = __builtin_amdgcn_wmma_f32_16x16x32_bf16(false, ahi.v, false, b1lo.v, (short)0, cmx1, false, false);
        cmx0 = __builtin_amdgcn_wmma_f32_16x16x32_bf16(false, alo.v, false, b0hi.v, (short)0, cmx0, false, false);
        cmx1 = __builtin_amdgcn_wmma_f32_16x16x32_bf16(false, alo.v, false, b1hi.v, (short)0, cmx1, false, false);
    }
    Tile2 t;
    t.c0 = chh0 + cmx0;
    t.c1 = chh1 + cmx1;
    return t;
}

// ---------------- histogram pass (coarse: refined==0, refined: refined==1) ----------------
__global__ void hist_kernel(const unsigned short* __restrict__ hi, const unsigned short* __restrict__ lo,
                            const float* __restrict__ sq,
                            const float* __restrict__ scal_f, const unsigned* __restrict__ scal_u,
                            unsigned* __restrict__ ghist, int refined) {
    __shared__ unsigned lhist[NBINS];
    for (int i = threadIdx.x; i < NBINS; i += blockDim.x) lhist[i] = 0u;
    __syncthreads();

    const int wave = threadIdx.x >> 5;
    const int lane = threadIdx.x & 31;
    const int ptile = blockIdx.x * WPB + wave;

    float rlo, rw;
    if (refined) { rlo = scal_f[1]; rw = scal_f[2]; }
    else         { rlo = 0.f; rw = 4.0f * __uint_as_float(scal_u[0]); } // D <= (|x|+|y|)^2 <= 4*max||x||^2
    const float scale = (float)NBINS / rw;

    if (ptile < PTILES) {
        const int ti  = ptile / (TILES / 2);
        const int tj0 = (ptile % (TILES / 2)) * 2;
        const int ibase = ti * 16;
        Tile2 t = gram_tile2(hi, lo, ibase, tj0 * 16, lane);
        const int n = lane & 15, h = lane >> 4;
#pragma unroll
        for (int sub = 0; sub < 2; ++sub) {
            const int tj = tj0 + sub;
            const int jbase = tj * 16;
            const float sj = sq[jbase + n];
            const v8f& c = sub ? t.c1 : t.c0;
#pragma unroll
            for (int r = 0; r < 8; ++r) {
                const int mm = r + h * 8;               // C layout: VGPR r -> M = r + 8*(lane>>4)
                float d = fmaxf(sq[ibase + mm] + sj - 2.0f * c[r], 0.f);
                if (ti == tj && mm == n) d = 0.f;       // zeroed diagonal
                if (refined) {
                    int b = (int)floorf((d - rlo) * scale);
                    if (b >= 0 && b < NBINS) atomicAdd(&lhist[b], 1u);
                } else {
                    int b = (int)(d * scale);
                    b = b < 0 ? 0 : (b >= NBINS ? NBINS - 1 : b);
                    atomicAdd(&lhist[b], 1u);
                }
            }
        }
    }
    __syncthreads();
    for (int i = threadIdx.x; i < NBINS; i += blockDim.x)
        if (lhist[i]) atomicAdd(&ghist[i], lhist[i]);
}

// rank of the median element in the ascending sort of all N^2 values
// idx = N + (N*N - N - 1) / 2   (0-based)
__device__ __forceinline__ unsigned long long median_rank() {
    const unsigned long long N = NTOT;
    return N + (N * N - N - 1ull) / 2ull;
}

__global__ void select1_kernel(const unsigned* __restrict__ h1,
                               float* __restrict__ scal_f, unsigned* __restrict__ scal_u) {
    const float bound = 4.0f * __uint_as_float(scal_u[0]);
    const float w = bound / (float)NBINS;
    const unsigned long long RANK = median_rank();
    unsigned long long cum = 0; int b = NBINS - 1; unsigned long long below = 0;
    for (int i = 0; i < NBINS; ++i) {
        if (cum + (unsigned long long)h1[i] > RANK) { b = i; below = cum; break; }
        cum += h1[i];
        if (i == NBINS - 1) below = cum - h1[i];
    }
    scal_f[1] = (float)b * w;                     // refined range lo
    scal_f[2] = w;                                // refined range width
    scal_u[3] = (unsigned)(RANK - below);         // rank within refined range
}

__global__ void select2_kernel(const unsigned* __restrict__ h2,
                               float* __restrict__ scal_f, const unsigned* __restrict__ scal_u) {
    const float rlo = scal_f[1], w = scal_f[2];
    const unsigned rank2 = scal_u[3];
    unsigned long long cum = 0; int b = NBINS - 1;
    for (int i = 0; i < NBINS; ++i) {
        if (cum + (unsigned long long)h2[i] > (unsigned long long)rank2) { b = i; break; }
        cum += h2[i];
    }
    float bw = rlo + ((float)b + 0.5f) * (w / (float)NBINS);
    if (bw == 0.f) bw = 1.f;
    scal_f[4] = bw;
}

// ---------------- final pass: multi-kernel MMD block sums ----------------
__global__ void mmd_kernel(const unsigned short* __restrict__ hi, const unsigned short* __restrict__ lo,
                           const float* __restrict__ sq, const float* __restrict__ scal_f,
                           double* __restrict__ sums) {
    const int wave = threadIdx.x >> 5;
    const int lane = threadIdx.x & 31;
    const int ptile = blockIdx.x * WPB + wave;
    if (ptile >= PTILES) return;
    const int ti  = ptile / (TILES / 2);
    const int tj0 = (ptile % (TILES / 2)) * 2;
    const int XT = NX / 16;                        // 32 tiles cover the X block (even -> pair-uniform)
    if (ti >= XT && tj0 < XT) return;              // whole pair is in the unused YX block

    const int ibase = ti * 16;
    Tile2 t = gram_tile2(hi, lo, ibase, tj0 * 16, lane);

    const float inv2bw = 1.0f / (2.0f * scal_f[4]);
    const int n = lane & 15, h = lane >> 4;
    float s[3] = {0.f, 0.f, 0.f};
#pragma unroll
    for (int sub = 0; sub < 2; ++sub) {
        const int tj = tj0 + sub;
        int region;
        if      (ti <  XT && tj <  XT) region = 0; // K_XX
        else if (ti >= XT && tj >= XT) region = 1; // K_YY
        else if (ti <  XT && tj >= XT) region = 2; // K_XY (upper-right block only)
        else continue;                             // YX block not used by reference
        const int jbase = tj * 16;
        const float sj = sq[jbase + n];
        const v8f& c = sub ? t.c1 : t.c0;
        float acc = 0.f;
#pragma unroll
        for (int r = 0; r < 8; ++r) {
            const int mm = r + h * 8;
            if (ti == tj && mm == n) continue;     // exclude diagonal (trace-subtraction)
            const float d = fmaxf(sq[ibase + mm] + sj - 2.0f * c[r], 0.f);
            const float x = d * inv2bw;
            float k = 0.f, mul = 1.f;
#pragma unroll
            for (int i = 0; i < KNUM; ++i) { k += __expf(-x * mul); mul *= 2.f; }
            acc += k;
        }
        s[region] += acc;
    }
    // wave32 reduction of all three region sums
#pragma unroll
    for (int rgn = 0; rgn < 3; ++rgn) {
        float v = s[rgn];
        for (int off = 16; off > 0; off >>= 1) v += __shfl_xor(v, off, 32);
        if (lane == 0 && v != 0.f) atomicAdd(&sums[rgn], (double)v);
    }
}

__global__ void finalize_kernel(const double* __restrict__ sums, float* __restrict__ out) {
    const double kxx = sums[0] / ((double)NX * (double)(NX - 1));
    const double kyy = sums[1] / ((double)NY * (double)(NY - 1));
    const double kxy = sums[2] / ((double)NX * (double)NY);
    double m2 = kxx + kyy - 2.0 * kxy;
    if (m2 < 0.0) m2 = 0.0;
    out[0] = (float)sqrt(m2);
}

// ---------------- launch ----------------
extern "C" void kernel_launch(void* const* d_in, const int* in_sizes, int n_in,
                              void* d_out, int out_size, void* d_ws, size_t ws_size,
                              hipStream_t stream) {
    const float* X = (const float*)d_in[0];   // compressed_tokens (8,64,768) f32
    const float* Y = (const float*)d_in[1];   // original_tokens  (8,576,768) f32
    float* out = (float*)d_out;

    // workspace layout (bytes) — total ~15.8 MB
    char* ws = (char*)d_ws;
    size_t off = 0;
    unsigned short* hi = (unsigned short*)(ws + off); off += (size_t)NTOT * DDIM * 2;  // 7,864,320
    unsigned short* lo = (unsigned short*)(ws + off); off += (size_t)NTOT * DDIM * 2;  // 7,864,320
    float*    sq   = (float*)(ws + off);    off += (size_t)NTOT * 4;                   // 20,480
    unsigned* h1   = (unsigned*)(ws + off); off += (size_t)NBINS * 4;                  // 16,384
    unsigned* h2   = (unsigned*)(ws + off); off += (size_t)NBINS * 4;                  // 16,384
    double*   sums = (double*)(ws + off);   off += 3 * 8;
    unsigned* scal_u = (unsigned*)(ws + off);
    float*    scal_f = (float*)(ws + off);

    const int hblocks = PTILES / WPB;  // 6400 blocks x 8 waves = 51200 pair-tiles

    init_kernel<<<1, 256, 0, stream>>>(h1, h2, sums, scal_u);
    prep_kernel<<<NTOT, 256, 0, stream>>>(X, Y, hi, lo, sq, scal_u);
    hist_kernel<<<hblocks, 256, 0, stream>>>(hi, lo, sq, scal_f, scal_u, h1, 0);
    select1_kernel<<<1, 1, 0, stream>>>(h1, scal_f, scal_u);
    hist_kernel<<<hblocks, 256, 0, stream>>>(hi, lo, sq, scal_f, scal_u, h2, 1);
    select2_kernel<<<1, 1, 0, stream>>>(h2, scal_f, scal_u);
    mmd_kernel<<<hblocks, 256, 0, stream>>>(hi, lo, sq, scal_f, sums);
    finalize_kernel<<<1, 1, 0, stream>>>(sums, out);
}
